// NewsGNN_25649544691769
// MI455X (gfx1250) — compile-verified, matrix operand order checked by reading, not run on hardware
//
#include <hip/hip_runtime.h>

// ---------------------------------------------------------------------------
// MI455X (gfx1250) implementation: bf16 WMMA GEMMs + atomic edge softmax/agg.
// All kernels launched on `stream`; scratch lives in d_ws; no sync APIs used.
// ---------------------------------------------------------------------------

typedef __attribute__((ext_vector_type(16))) __bf16 v16bf;
typedef __attribute__((ext_vector_type(8)))  __bf16 v8bf;
typedef __attribute__((ext_vector_type(8)))  float  v8f;

__device__ __forceinline__ __bf16 tobf(float x) { return (__bf16)x; }

// Monotonic float <-> uint encoding for atomicMax-based segment max.
__device__ __forceinline__ unsigned f32_key(float f) {
  unsigned u = __float_as_uint(f);
  return (u & 0x80000000u) ? ~u : (u | 0x80000000u);
}
__device__ __forceinline__ float key_f32(unsigned k) {
  unsigned u = (k & 0x80000000u) ? (k & 0x7FFFFFFFu) : ~k;
  return __uint_as_float(u);
}

// ---------------------------------------------------------------------------
// Tiled GEMM: C[M,N] = act(A[M,K]_f32 @ Bt[N,K]_bf16^T + bias)
// Block = 256 threads = 8 waves; block tile 64(M) x 32(N); wave tile 16x16.
// K, N multiples of 32. A rows out of range are clamped to row 0 (loads stay
// valid; garbage accumulators map only to store-guarded rows), so the K-loop
// has NO divergence: loads + cvt_pk_bf16 + v_wmma only, EXEC stays all-ones.
// ---------------------------------------------------------------------------
__global__ __launch_bounds__(256)
void gemm_wmma_bf16(const float* __restrict__ A, const __bf16* __restrict__ Bt,
                    const float* __restrict__ bias, float* __restrict__ C,
                    int M, int K, int N, int act /*0=none,1=relu*/) {
  const int tid  = threadIdx.x;
  const int wave = tid >> 5;
  const int lane = tid & 31;
  const int wm   = wave >> 1;    // 0..3
  const int wn   = wave & 1;     // 0..1
  const int half = lane >> 4;    // lane group
  const int ml   = lane & 15;

  const int baseM = blockIdx.y * 64 + wm * 16;
  const int baseN = blockIdx.x * 32 + wn * 16;

  int arowIdx = baseM + ml;                    // A-fragment: M = lane&15
  if (arowIdx >= M) arowIdx = 0;               // clamp: loads valid, store-guarded
  const float*  arow = A  + (size_t)arowIdx * K;
  const __bf16* brow = Bt + (size_t)(baseN + ml) * K;   // B-fragment: N = lane&15

  v8f acc = {};
  for (int k = 0; k < K; k += 32) {
    const int k0 = k + half * 8;        // K pairs for VGPRs 0..3 of this half
    const int k1 = k + 16 + half * 8;   // K pairs for VGPRs 4..7

    const float4 f0 = *(const float4*)(arow + k0);
    const float4 f1 = *(const float4*)(arow + k0 + 4);
    const float4 f2 = *(const float4*)(arow + k1);
    const float4 f3 = *(const float4*)(arow + k1 + 4);
    if (k + 32 < K) __builtin_prefetch(arow + k + 32 + half * 8, 0, 1);

    v16bf a;
    a[0]  = tobf(f0.x); a[1]  = tobf(f0.y); a[2]  = tobf(f0.z); a[3]  = tobf(f0.w);
    a[4]  = tobf(f1.x); a[5]  = tobf(f1.y); a[6]  = tobf(f1.z); a[7]  = tobf(f1.w);
    a[8]  = tobf(f2.x); a[9]  = tobf(f2.y); a[10] = tobf(f2.z); a[11] = tobf(f2.w);
    a[12] = tobf(f3.x); a[13] = tobf(f3.y); a[14] = tobf(f3.z); a[15] = tobf(f3.w);

    const v8bf b0 = *(const v8bf*)(brow + k0);
    const v8bf b1 = *(const v8bf*)(brow + k1);
    v16bf b = __builtin_shufflevector(b0, b1, 0,1,2,3,4,5,6,7,8,9,10,11,12,13,14,15);

    acc = __builtin_amdgcn_wmma_f32_16x16x32_bf16(false, a, false, b,
                                                  (short)0, acc, false, false);
  }

  const int   col = baseN + ml;
  const float bv  = bias ? bias[col] : 0.0f;
#pragma unroll
  for (int r = 0; r < 8; ++r) {
    const int mrow = baseM + 8 * half + r;   // C layout: VGPR r -> M = r + 8*half
    if (mrow < M) {
      float v = acc[r] + bv;
      if (act == 1) v = fmaxf(v, 0.0f);
      C[(size_t)mrow * N + col] = v;
    }
  }
}

// Convert + transpose weight: W[K,N] f32 -> Wt[N,K] bf16
__global__ void convert_wt(const float* __restrict__ W, __bf16* __restrict__ Wt,
                           int K, int N) {
  int i = blockIdx.x * blockDim.x + threadIdx.x;
  if (i >= K * N) return;
  int k = i / N, n = i - k * N;
  Wt[(size_t)n * K + k] = tobf(W[i]);
}

__global__ void fill_u32(unsigned* __restrict__ p, unsigned v, long n) {
  long i = (long)blockIdx.x * blockDim.x + threadIdx.x;
  if (i < n) p[i] = v;
}

// a_src / a_dst: per (node, head) dot of h[n,head,:] with attention vectors.
__global__ void attn_scores(const float* __restrict__ h,
                            const float* __restrict__ att_s,
                            const float* __restrict__ att_d,
                            float* __restrict__ as_, float* __restrict__ ad_,
                            int N4) {
  int i = blockIdx.x * blockDim.x + threadIdx.x;
  if (i >= N4) return;
  const int head = i & 3;
  const float* hp = h + (size_t)i * 32;
  const float* sp = att_s + head * 32;
  const float* dp = att_d + head * 32;
  float s = 0.f, d = 0.f;
#pragma unroll 8
  for (int j = 0; j < 32; ++j) { float v = hp[j]; s = fmaf(v, sp[j], s); d = fmaf(v, dp[j], d); }
  as_[i] = s; ad_[i] = d;
}

__device__ __forceinline__ void edge_sd(const int* __restrict__ ei, int E, int e,
                                        int& s, int& d) {
  if (e < E) { s = ei[e]; d = ei[E + e]; } else { s = e - E; d = s; }
}

// Pass 1: segment max of leaky_relu(a_src[src]+a_dst[dst]) over dst, per head.
__global__ void edge_max(const int* __restrict__ ei, int E, int N,
                         const float* __restrict__ as_, const float* __restrict__ ad_,
                         unsigned* __restrict__ mkey) {
  int e = blockIdx.x * blockDim.x + threadIdx.x;
  if (e >= E + N) return;
  int s, d; edge_sd(ei, E, e, s, d);
#pragma unroll
  for (int h = 0; h < 4; ++h) {
    float v = as_[s * 4 + h] + ad_[d * 4 + h];
    v = v > 0.f ? v : 0.2f * v;
    atomicMax(&mkey[d * 4 + h], f32_key(v));
  }
}

// Pass 2: p = exp(e - m[dst]); denom[dst] += p; store p per (edge, head).
__global__ void edge_denom(const int* __restrict__ ei, int E, int N,
                           const float* __restrict__ as_, const float* __restrict__ ad_,
                           const unsigned* __restrict__ mkey,
                           float* __restrict__ denom, float* __restrict__ pbuf) {
  int e = blockIdx.x * blockDim.x + threadIdx.x;
  if (e >= E + N) return;
  int s, d; edge_sd(ei, E, e, s, d);
#pragma unroll
  for (int h = 0; h < 4; ++h) {
    float v = as_[s * 4 + h] + ad_[d * 4 + h];
    v = v > 0.f ? v : 0.2f * v;
    float p = __expf(v - key_f32(mkey[d * 4 + h]));
    pbuf[(size_t)e * 4 + h] = p;
    atomicAdd(&denom[d * 4 + h], p);
  }
}

// Pass 3: agg[dst,f] += (p/denom) * h[src,f]; 128 threads per edge.
__global__ __launch_bounds__(256)
void edge_aggregate(const int* __restrict__ ei, int E, int N,
                    const float* __restrict__ pbuf, const float* __restrict__ denom,
                    const float* __restrict__ h, float* __restrict__ agg) {
  int t = blockIdx.x * 256 + threadIdx.x;
  int e = t >> 7;
  if (e >= E + N) return;
  int f = t & 127;
  int s, d; edge_sd(ei, E, e, s, d);
  int head = f >> 5;
  float alpha = pbuf[(size_t)e * 4 + head] / (denom[d * 4 + head] + 1e-16f);
  atomicAdd(&agg[(size_t)d * 128 + f], alpha * h[(size_t)s * 128 + f]);
}

// x_out = elu(agg + bias)
__global__ void bias_elu(const float* __restrict__ agg, const float* __restrict__ bias,
                         float* __restrict__ xout, long n) {
  long i = (long)blockIdx.x * blockDim.x + threadIdx.x;
  if (i >= n) return;
  float v = agg[i] + bias[i & 127];
  xout[i] = v > 0.f ? v : (__expf(v) - 1.0f);
}

// Final tiny projection: out[r,0:2] = c1[r,:] @ Wc2 + bc2
__global__ void classifier2(const float* __restrict__ c1, const float* __restrict__ Wc2,
                            const float* __restrict__ bc2, float* __restrict__ out, int M) {
  int r = blockIdx.x * blockDim.x + threadIdx.x;
  if (r >= M) return;
  float o0 = bc2[0], o1 = bc2[1];
  const float* row = c1 + (size_t)r * 64;
#pragma unroll 8
  for (int j = 0; j < 64; ++j) {
    float v = row[j];
    o0 = fmaf(v, Wc2[j * 2 + 0], o0);
    o1 = fmaf(v, Wc2[j * 2 + 1], o1);
  }
  out[(size_t)r * 2 + 0] = o0;
  out[(size_t)r * 2 + 1] = o1;
}

// ---------------------------------------------------------------------------
extern "C" void kernel_launch(void* const* d_in, const int* in_sizes, int n_in,
                              void* d_out, int out_size, void* d_ws, size_t ws_size,
                              hipStream_t stream) {
  const float* x_news = (const float*)d_in[0];
  const float* x_tw   = (const float*)d_in[1];
  const int*   ei     = (const int*)d_in[2];
  const float* W_news = (const float*)d_in[3];
  const float* b_news = (const float*)d_in[4];
  const float* W_tw   = (const float*)d_in[5];
  const float* b_tw   = (const float*)d_in[6];
  const float* gW[2]  = {(const float*)d_in[7],  (const float*)d_in[11]};
  const float* gas[2] = {(const float*)d_in[8],  (const float*)d_in[12]};
  const float* gad[2] = {(const float*)d_in[9],  (const float*)d_in[13]};
  const float* gb[2]  = {(const float*)d_in[10], (const float*)d_in[14]};
  const float* Wc1 = (const float*)d_in[15];
  const float* bc1 = (const float*)d_in[16];
  const float* Wc2 = (const float*)d_in[17];
  const float* bc2 = (const float*)d_in[18];
  (void)n_in; (void)out_size; (void)ws_size;

  const long n_news = in_sizes[0] / 768;
  const long n_tw   = in_sizes[1] / 512;
  const long N      = n_news + n_tw;
  const long E      = in_sizes[2] / 2;
  const long Etot   = E + N;

  // Workspace carve-out (256B aligned).
  char* w = (char*)d_ws;
  auto carve = [&](size_t bytes) -> char* {
    char* p = w; w += (bytes + 255) & ~(size_t)255; return p;
  };
  float*    xbuf  = (float*)carve((size_t)N * 128 * 4);
  float*    hbuf  = (float*)carve((size_t)N * 128 * 4);
  float*    agg   = (float*)carve((size_t)N * 128 * 4);
  float*    asb   = (float*)carve((size_t)N * 4 * 4);
  float*    adb   = (float*)carve((size_t)N * 4 * 4);
  unsigned* mkey  = (unsigned*)carve((size_t)N * 4 * 4);
  float*    denom = (float*)carve((size_t)N * 4 * 4);
  float*    pbuf  = (float*)carve((size_t)Etot * 4 * 4);
  float*    c1buf = (float*)carve((size_t)n_news * 64 * 4);
  __bf16*   wtN   = (__bf16*)carve(768 * 128 * 2);
  __bf16*   wtT   = (__bf16*)carve(512 * 128 * 2);
  __bf16*   wtG[2]= {(__bf16*)carve(128 * 128 * 2), (__bf16*)carve(128 * 128 * 2)};
  __bf16*   wtC1  = (__bf16*)carve(128 * 64 * 2);

  auto cdiv = [](long a, long b) { return (unsigned)((a + b - 1) / b); };

  // Pre-convert + transpose weights to bf16 [N][K].
  convert_wt<<<cdiv(768 * 128, 256), 256, 0, stream>>>(W_news, wtN, 768, 128);
  convert_wt<<<cdiv(512 * 128, 256), 256, 0, stream>>>(W_tw,   wtT, 512, 128);
  convert_wt<<<cdiv(128 * 128, 256), 256, 0, stream>>>(gW[0],  wtG[0], 128, 128);
  convert_wt<<<cdiv(128 * 128, 256), 256, 0, stream>>>(gW[1],  wtG[1], 128, 128);
  convert_wt<<<cdiv(128 * 64,  256), 256, 0, stream>>>(Wc1,    wtC1, 128, 64);

  auto gemm = [&](const float* A, const __bf16* Bt, const float* bias, float* C,
                  long M, int K, int Nc, int act) {
    dim3 grid(Nc / 32, cdiv(M, 64));
    gemm_wmma_bf16<<<grid, 256, 0, stream>>>(A, Bt, bias, C, (int)M, K, Nc, act);
  };

  // Feature encoders -> xbuf [N,128]
  gemm(x_news, wtN, b_news, xbuf, n_news, 768, 128, 1);
  gemm(x_tw,   wtT, b_tw,   xbuf + (size_t)n_news * 128, n_tw, 512, 128, 1);

  // Two GAT layers
  for (int layer = 0; layer < 2; ++layer) {
    gemm(xbuf, wtG[layer], nullptr, hbuf, N, 128, 128, 0);
    attn_scores<<<cdiv(N * 4, 256), 256, 0, stream>>>(hbuf, gas[layer], gad[layer],
                                                      asb, adb, (int)(N * 4));
    fill_u32<<<cdiv(N * 128, 256), 256, 0, stream>>>((unsigned*)agg, 0u, N * 128);
    fill_u32<<<cdiv(N * 4, 256), 256, 0, stream>>>(mkey, 0u, N * 4);
    fill_u32<<<cdiv(N * 4, 256), 256, 0, stream>>>((unsigned*)denom, 0u, N * 4);

    edge_max<<<cdiv(Etot, 256), 256, 0, stream>>>(ei, (int)E, (int)N, asb, adb, mkey);
    edge_denom<<<cdiv(Etot, 256), 256, 0, stream>>>(ei, (int)E, (int)N, asb, adb,
                                                    mkey, denom, pbuf);
    edge_aggregate<<<cdiv(Etot * 128, 256), 256, 0, stream>>>(ei, (int)E, (int)N,
                                                              pbuf, denom, hbuf, agg);
    bias_elu<<<cdiv(N * 128, 256), 256, 0, stream>>>(agg, gb[layer], xbuf, N * 128);
  }

  // Classifier head (only news rows).
  gemm(xbuf, wtC1, bc1, c1buf, n_news, 128, 64, 1);
  classifier2<<<cdiv(n_news, 256), 256, 0, stream>>>(c1buf, Wc2, bc2, (float*)d_out,
                                                     (int)n_news);
}
